// ALTE_7645041787124
// MI455X (gfx1250) — matrix operand order repaired and checked by reading, not compiled, not run on hardware
//
#include <hip/hip_runtime.h>
#include <hip/hip_bf16.h>

typedef _Float16 half_t;
typedef half_t v16h __attribute__((ext_vector_type(16)));
typedef half_t v8h  __attribute__((ext_vector_type(8)));
typedef float  v8f  __attribute__((ext_vector_type(8)));

#define HW      128
#define RS      264                    // padded LDS row stride (f16 elems) to dodge bank conflicts
#define ACT_B   (16 * RS * 2)          // 8448 B : one 16x256 f16 activation tile
#define F32_B   (16 * RS * 4)          // 16896 B: one 16x256 f32 tile
#define WAVE_B  (2 * ACT_B + F32_B)    // 33792 B per wave
#define MAIN_LDS (8 * WAVE_B)          // 270336 B per 8-wave block (fits 320KB WGP LDS)
#define CF_LDS  (18 * 18 * 64 * 4)     // 82944 B feat tile for the 3x3 convs

// ---------------------------------------------------------------------------
// Weight repack: W (N x K, f32, row-major) -> f16 WMMA B-fragments.
// Fragment layout: frag[((nt*(K/32)+kt)*32 + lane)*16 + e] = W[nt*16+(lane&15)][kt*32+(lane>>4)*8+(e&7)+(e>>3)*16]
// so each lane's 16 f16 (32B) for one v_wmma B operand are contiguous.
// ---------------------------------------------------------------------------
__global__ void makefrag_kernel(const float* __restrict__ W, half_t* __restrict__ dst,
                                int N, int K) {
    int i = blockIdx.x * blockDim.x + threadIdx.x;
    if (i >= N * K) return;
    int e    = i & 15;
    int lane = (i >> 4) & 31;
    int rest = i >> 9;
    int ktn  = K / 32;
    int kt   = rest % ktn;
    int nt   = rest / ktn;
    int n = nt * 16 + (lane & 15);
    int k = kt * 32 + (lane >> 4) * 8 + (e & 7) + (e >> 3) * 16;
    dst[i] = (half_t)W[n * K + k];
}

// ---------------------------------------------------------------------------
// enc conv: (3,128,128) f32 -> feat HWC (128*128, 64) f32
// ---------------------------------------------------------------------------
__global__ void conv_enc_kernel(const float* __restrict__ inp, const float* __restrict__ w,
                                const float* __restrict__ b, float* __restrict__ featH) {
    int pix = blockIdx.x * blockDim.x + threadIdx.x;
    if (pix >= HW * HW) return;
    int y = pix >> 7, x = pix & 127;
    float patch[27];
    for (int ic = 0; ic < 3; ++ic)
        for (int ky = 0; ky < 3; ++ky)
            for (int kx = 0; kx < 3; ++kx) {
                int yy = y + ky - 1, xx = x + kx - 1;
                patch[ic * 9 + ky * 3 + kx] =
                    (yy >= 0 && yy < HW && xx >= 0 && xx < HW) ? inp[ic * HW * HW + yy * HW + xx] : 0.f;
            }
    for (int oc = 0; oc < 64; ++oc) {
        float s = b[oc];
        const float* wp = w + oc * 27;
#pragma unroll
        for (int i = 0; i < 27; ++i) s += patch[i] * wp[i];
        featH[(size_t)pix * 64 + oc] = s;
    }
}

// ---------------------------------------------------------------------------
// coef/freq convs: feat HWC f32 -> coef/freq HWC f16 (64 -> 256 ch, 3x3)
// 16x16 pixel tile per block, 18x18x64 feat tile in LDS.
// ---------------------------------------------------------------------------
__global__ void conv_cf_kernel(const float* __restrict__ featH,
                               const float* __restrict__ cw, const float* __restrict__ cb,
                               const float* __restrict__ fw, const float* __restrict__ fb,
                               half_t* __restrict__ coefH, half_t* __restrict__ freqH) {
    extern __shared__ float tile[];
    int by = blockIdx.x >> 3, bx = blockIdx.x & 7;
    int y0 = by * 16, x0 = bx * 16;
    for (int i = threadIdx.x; i < 18 * 18 * 64; i += 256) {
        int ic = i & 63;
        int rest = i >> 6;
        int tx = rest % 18, ty = rest / 18;
        int yy = y0 + ty - 1, xx = x0 + tx - 1;
        tile[i] = (yy >= 0 && yy < HW && xx >= 0 && xx < HW)
                    ? featH[((size_t)(yy * HW + xx)) * 64 + ic] : 0.f;
    }
    __syncthreads();
    int py = threadIdx.x >> 4, px = threadIdx.x & 15;
    size_t pix = (size_t)(y0 + py) * HW + (x0 + px);
    for (int oc = 0; oc < 256; oc += 4) {
        float ac0 = cb[oc], ac1 = cb[oc + 1], ac2 = cb[oc + 2], ac3 = cb[oc + 3];
        float af0 = fb[oc], af1 = fb[oc + 1], af2 = fb[oc + 2], af3 = fb[oc + 3];
        for (int t = 0; t < 9; ++t) {
            int ky = t / 3, kx = t % 3;
            const float* tp  = tile + ((py + ky) * 18 + (px + kx)) * 64;
            const float* cwp = cw + (size_t)oc * 576 + t;
            const float* fwp = fw + (size_t)oc * 576 + t;
#pragma unroll 8
            for (int ic = 0; ic < 64; ++ic) {
                float fv = tp[ic];
                size_t wi = (size_t)ic * 9;
                ac0 += fv * cwp[wi];        ac1 += fv * cwp[wi + 576];
                ac2 += fv * cwp[wi + 1152]; ac3 += fv * cwp[wi + 1728];
                af0 += fv * fwp[wi];        af1 += fv * fwp[wi + 576];
                af2 += fv * fwp[wi + 1152]; af3 += fv * fwp[wi + 1728];
            }
        }
        half_t* cp = coefH + pix * 256 + oc;
        cp[0] = (half_t)ac0; cp[1] = (half_t)ac1; cp[2] = (half_t)ac2; cp[3] = (half_t)ac3;
        half_t* fp = freqH + pix * 256 + oc;
        fp[0] = (half_t)af0; fp[1] = (half_t)af1; fp[2] = (half_t)af2; fp[3] = (half_t)af3;
    }
}

// ---------------------------------------------------------------------------
// WMMA helpers (wave32)
// ---------------------------------------------------------------------------
__device__ __forceinline__ v16h load_a_frag(const half_t* A, int lane, int kbase) {
    // ISA 16-bit A layout: lanes 0-15: K = kbase+{0..7, 16..23}; lanes 16-31: +8
    int m = lane & 15, h = lane >> 4;
    const half_t* p = A + m * RS + kbase + h * 8;
    union { v16h v; v8h h2[2]; } u;
    u.h2[0] = *(const v8h*)(p);
    u.h2[1] = *(const v8h*)(p + 16);
    return u.v;
}

__device__ __forceinline__ v16h load_b_frag(const half_t* frag, int lane, int idx) {
    const half_t* p = frag + ((size_t)idx * 32 + lane) * 16;
    union { v16h v; v8h h2[2]; } u;
    u.h2[0] = *(const v8h*)(p);
    u.h2[1] = *(const v8h*)(p + 8);
    return u.v;
}

// C(16x256) = A(16xK_local) @ W  (W fragment, ktnW = K_total/32, starting at kt0)
// mode 0: dst16[m][n] = (relu?)(acc + bias[n])   (f16)
// mode 1: dst32[m][n] += acc                      (f32 accumulate, no bias)
// mode 2: dst32[m][n] = acc + bias[n]             (f32)
__device__ __forceinline__ void wave_gemm(const half_t* A, const half_t* Wfrag,
                                          int ktnW, int kt0, int kloc,
                                          const float* bias,
                                          half_t* dst16, float* dst32,
                                          int lane, bool relu, int mode) {
    for (int nt = 0; nt < 16; ++nt) {
        v8f acc = {};
        for (int kt = 0; kt < kloc; ++kt) {
            v16h a = load_a_frag(A, lane, kt * 32);
            v16h b = load_b_frag(Wfrag, lane, nt * ktnW + kt0 + kt);
            acc = __builtin_amdgcn_wmma_f32_16x16x32_f16(false, a, false, b,
                                                         (short)0, acc, false, false);
        }
        int n = nt * 16 + (lane & 15);
        int mbase = (lane >> 4) * 8;
        float bv = bias ? bias[n] : 0.f;
#pragma unroll
        for (int r = 0; r < 8; ++r) {
            int m = mbase + r;
            if (mode == 1) {
                dst32[m * RS + n] += acc[r];
            } else {
                float v = acc[r] + bv;
                if (relu) v = fmaxf(v, 0.f);
                if (mode == 0) dst16[m * RS + n] = (half_t)v;
                else           dst32[m * RS + n] = v;
            }
        }
    }
}

__device__ __forceinline__ void corner_geom(float cy, float cx, int k,
                                            int& iy, int& ix, float& rel0, float& rel1) {
    const float r = 1.f / 128.f, eps = 1e-6f;
    float vx = (k & 2) ? 1.f : -1.f;
    float vy = (k & 1) ? 1.f : -1.f;
    float c0 = fminf(fmaxf(cy + vx * r + eps, -1.f + 1e-6f), 1.f - 1e-6f);
    float c1 = fminf(fmaxf(cx + vy * r + eps, -1.f + 1e-6f), 1.f - 1e-6f);
    float fy = ((c0 + 1.f) * 128.f - 1.f) * 0.5f;
    float fx = ((c1 + 1.f) * 128.f - 1.f) * 0.5f;
    iy = min(max((int)roundf(fy), 0), 127);
    ix = min(max((int)roundf(fx), 0), 127);
    float qcy = -1.f + (2.f * iy + 1.f) / 128.f;
    float qcx = -1.f + (2.f * ix + 1.f) / 128.f;
    rel0 = (cy - qcy) * 128.f;
    rel1 = (cx - qcx) * 128.f;
}

// ---------------------------------------------------------------------------
// Fused per-query kernel: 8 waves/block, each wave owns 16 queries end-to-end.
// ---------------------------------------------------------------------------
__global__ void __launch_bounds__(256, 1)
main_kernel(const float* __restrict__ coord, const float* __restrict__ cell,
            const float* __restrict__ inp,
            const half_t* __restrict__ coefH, const half_t* __restrict__ freqH,
            const half_t* __restrict__ pinF, const float* __restrict__ pin_b,
            const half_t* __restrict__ m1F,  const float* __restrict__ m1_b,
            const half_t* __restrict__ m2F,  const float* __restrict__ m2_b,
            const half_t* __restrict__ poutF,const float* __restrict__ pout_b,
            const half_t* __restrict__ im1F, const float* __restrict__ im1_b,
            const half_t* __restrict__ im2F, const float* __restrict__ im2_b,
            const float* __restrict__ im3_w, const float* __restrict__ im3_b,
            const float* __restrict__ phase_w, float* __restrict__ out) {
    extern __shared__ char smem[];
    int wave = threadIdx.x >> 5;
    int lane = threadIdx.x & 31;
    char* wbase = smem + (size_t)wave * WAVE_B;
    half_t* act  = (half_t*)wbase;                // 16x256 f16 (t1 / wx / y1 ...)
    half_t* hdd  = (half_t*)(wbase + ACT_B);      // 16x256 f16
    float*  f32b = (float*)(wbase + 2 * ACT_B);   // 16x256 f32 (hdd accum / aw)
    int qbase = (blockIdx.x * 8 + wave) * 16;

    int m = lane & 15, h = lane >> 4;
    int q = qbase + m;
    float cy = coord[q * 2 + 0], cx = coord[q * 2 + 1];
    float rc0 = cell[q * 2 + 0] * 128.f, rc1 = cell[q * 2 + 1] * 128.f;

    // areas for all 4 corners (needed before weighting)
    float areas[4], tot = 0.f;
#pragma unroll
    for (int k = 0; k < 4; ++k) {
        int iy, ix; float r0, r1;
        corner_geom(cy, cx, k, iy, ix, r0, r1);
        areas[k] = fabsf(r0 * r1) + 1e-9f;
        tot += areas[k];
    }

    // zero hdd f32 accumulator
    for (int i = lane; i < 16 * RS; i += 32) f32b[i] = 0.f;
    __builtin_amdgcn_wave_barrier();

    const float PI = 3.14159265358979f;
    for (int k = 0; k < 4; ++k) {
        int iy, ix; float r0, r1;
        corner_geom(cy, cx, k, iy, ix, r0, r1);
        float wgt = areas[3 - k] / tot;            // reference reverses area order
        const half_t* qc = coefH + (size_t)(iy * HW + ix) * 256;
        const half_t* qf = freqH + (size_t)(iy * HW + ix) * 256;
        // lane (m,h): h=0 -> cos half (c=0..127), h=1 -> sin half (c=128..255)
        for (int j = 0; j < 128; ++j) {
            float f0 = (float)qf[2 * j], f1 = (float)qf[2 * j + 1];
            float ph = rc0 * phase_w[2 * j] + rc1 * phase_w[2 * j + 1];
            float f  = f0 * r0 + f1 * r1 + ph;
            float t  = PI * f;
            float ffv = h ? __sinf(t) : __cosf(t);
            int c = h * 128 + j;
            act[m * RS + c] = (half_t)((float)qc[c] * ffv * wgt);
        }
        __builtin_amdgcn_wave_barrier();
        // hdd += act @ pin_w_k^T   (pin fragment: ktnW=32, corner slice kt0 = k*8)
        wave_gemm(act, pinF, 32, k * 8, 8, nullptr, nullptr, f32b, lane, false, 1);
        __builtin_amdgcn_wave_barrier();
    }

    // hdd f16 = f32 accum + pin_b
    for (int i = lane; i < 16 * 256; i += 32) {
        int mm = i >> 8, c = i & 255;
        hdd[mm * RS + c] = (half_t)(f32b[mm * RS + c] + pin_b[c]);
    }
    __builtin_amdgcn_wave_barrier();

    // t1 = relu(hdd @ m1 + b) -> act
    wave_gemm(hdd, m1F, 8, 0, 8, m1_b, act, nullptr, lane, true, 0);
    __builtin_amdgcn_wave_barrier();
    // aw = t1 @ m2 + b -> f32b
    wave_gemm(act, m2F, 8, 0, 8, m2_b, nullptr, f32b, lane, false, 2);
    __builtin_amdgcn_wave_barrier();

    // per-head softmax (8 heads x 32) and wx = aw * hdd -> act
#pragma unroll
    for (int it = 0; it < 4; ++it) {
        int g = it * 32 + lane;
        int mm = g >> 3, hh = g & 7;
        const float* row = f32b + mm * RS + hh * 32;
        float mx = -1e30f;
        for (int i = 0; i < 32; ++i) mx = fmaxf(mx, row[i]);
        float s = 0.f;
        for (int i = 0; i < 32; ++i) s += __expf(row[i] - mx);
        float inv = 1.f / s;
        for (int i = 0; i < 32; ++i) {
            int c = hh * 32 + i;
            float awv = __expf(row[i] - mx) * inv;
            act[mm * RS + c] = (half_t)(awv * (float)hdd[mm * RS + c]);
        }
    }
    __builtin_amdgcn_wave_barrier();

    // attn = wx @ pout + b -> hdd ; y1 = relu(attn @ im1 + b) -> act ; y2 = relu(y1 @ im2 + b) -> hdd
    wave_gemm(act, poutF, 8, 0, 8, pout_b, hdd, nullptr, lane, false, 0);
    __builtin_amdgcn_wave_barrier();
    wave_gemm(hdd, im1F, 8, 0, 8, im1_b, act, nullptr, lane, true, 0);
    __builtin_amdgcn_wave_barrier();
    wave_gemm(act, im2F, 8, 0, 8, im2_b, hdd, nullptr, lane, true, 0);
    __builtin_amdgcn_wave_barrier();

    // y = y2 @ im3 + b + bilinear(inp, coord)
    for (int idx = lane; idx < 48; idx += 32) {
        int mm = idx / 3, c = idx % 3;
        float s = im3_b[c];
        const float* wrow = im3_w + c * 256;
        for (int kk = 0; kk < 256; ++kk) s += (float)hdd[mm * RS + kk] * wrow[kk];
        int qq = qbase + mm;
        float qy = coord[qq * 2 + 0], qx = coord[qq * 2 + 1];
        float fy = ((qy + 1.f) * 128.f - 1.f) * 0.5f;
        float fx = ((qx + 1.f) * 128.f - 1.f) * 0.5f;
        float y0f = floorf(fy), x0f = floorf(fx);
        float wy = fy - y0f, wx2 = fx - x0f;
        int y0 = min(max((int)y0f, 0), 127),     y1 = min(max((int)y0f + 1, 0), 127);
        int x0 = min(max((int)x0f, 0), 127),     x1 = min(max((int)x0f + 1, 0), 127);
        const float* ch = inp + c * HW * HW;
        float v00 = ch[y0 * HW + x0], v01 = ch[y0 * HW + x1];
        float v10 = ch[y1 * HW + x0], v11 = ch[y1 * HW + x1];
        float bil = v00 * (1.f - wy) * (1.f - wx2) + v01 * (1.f - wy) * wx2 +
                    v10 * wy * (1.f - wx2) + v11 * wy * wx2;
        out[(size_t)qq * 3 + c] = s + bil;
    }
}

// ---------------------------------------------------------------------------
// Host side
// ---------------------------------------------------------------------------
extern "C" void kernel_launch(void* const* d_in, const int* in_sizes, int n_in,
                              void* d_out, int out_size, void* d_ws, size_t ws_size,
                              hipStream_t stream) {
    const float* inp     = (const float*)d_in[0];
    const float* coord   = (const float*)d_in[1];
    const float* cell    = (const float*)d_in[2];
    const float* enc_w   = (const float*)d_in[3];
    const float* enc_b   = (const float*)d_in[4];
    const float* coef_w  = (const float*)d_in[5];
    const float* coef_b  = (const float*)d_in[6];
    const float* freq_w  = (const float*)d_in[7];
    const float* freq_b  = (const float*)d_in[8];
    const float* phase_w = (const float*)d_in[9];
    const float* pin_w   = (const float*)d_in[10];
    const float* pin_b   = (const float*)d_in[11];
    const float* m1_w    = (const float*)d_in[12];
    const float* m1_b    = (const float*)d_in[13];
    const float* m2_w    = (const float*)d_in[14];
    const float* m2_b    = (const float*)d_in[15];
    const float* pout_w  = (const float*)d_in[16];
    const float* pout_b  = (const float*)d_in[17];
    const float* im1_w   = (const float*)d_in[18];
    const float* im1_b   = (const float*)d_in[19];
    const float* im2_w   = (const float*)d_in[20];
    const float* im2_b   = (const float*)d_in[21];
    const float* im3_w   = (const float*)d_in[22];
    const float* im3_b   = (const float*)d_in[23];
    float* out = (float*)d_out;

    char* ws = (char*)d_ws;
    float*  featH = (float*)(ws + 0);                 //  4 MB
    half_t* coefH = (half_t*)(ws + 4194304);          //  8 MB
    half_t* freqH = (half_t*)(ws + 12582912);         //  8 MB
    half_t* pinF  = (half_t*)(ws + 20971520);         // 512 KB
    half_t* m1F   = (half_t*)(ws + 21495808);
    half_t* m2F   = (half_t*)(ws + 21626880);
    half_t* poutF = (half_t*)(ws + 21757952);
    half_t* im1F  = (half_t*)(ws + 21889024);
    half_t* im2F  = (half_t*)(ws + 22020096);

    (void)hipFuncSetAttribute((const void*)main_kernel,
                              hipFuncAttributeMaxDynamicSharedMemorySize, MAIN_LDS);
    (void)hipFuncSetAttribute((const void*)conv_cf_kernel,
                              hipFuncAttributeMaxDynamicSharedMemorySize, CF_LDS);

    // weight repack to WMMA fragments (f16)
    makefrag_kernel<<<(256 * 1024) / 256, 256, 0, stream>>>(pin_w, pinF, 256, 1024);
    makefrag_kernel<<<(256 * 256) / 256, 256, 0, stream>>>(m1_w,  m1F,  256, 256);
    makefrag_kernel<<<(256 * 256) / 256, 256, 0, stream>>>(m2_w,  m2F,  256, 256);
    makefrag_kernel<<<(256 * 256) / 256, 256, 0, stream>>>(pout_w, poutF, 256, 256);
    makefrag_kernel<<<(256 * 256) / 256, 256, 0, stream>>>(im1_w, im1F, 256, 256);
    makefrag_kernel<<<(256 * 256) / 256, 256, 0, stream>>>(im2_w, im2F, 256, 256);

    // convs
    conv_enc_kernel<<<(HW * HW) / 256, 256, 0, stream>>>(inp, enc_w, enc_b, featH);
    conv_cf_kernel<<<64, 256, CF_LDS, stream>>>(featH, coef_w, coef_b, freq_w, freq_b,
                                                coefH, freqH);

    // fused per-query WMMA chain
    int Q = in_sizes[1] / 2;          // 65536
    int blocks = Q / 128;             // 16 queries/wave * 8 waves
    main_kernel<<<blocks, 256, MAIN_LDS, stream>>>(coord, cell, inp, coefH, freqH,
        pinF, pin_b, m1F, m1_b, m2F, m2_b, poutF, pout_b,
        im1F, im1_b, im2F, im2_b, im3_w, im3_b, phase_w, out);
}
